// cape_cau_64733747085777
// MI455X (gfx1250) — matrix operand order repaired and checked by reading, not compiled, not run on hardware
//
#include <hip/hip_runtime.h>
#include <cstdint>

// ---------------- problem constants ----------------
#define B_    64
#define T_    8
#define N_    2000
#define F_    16
#define CH_   64
#define BN_   (B_ * N_)       // 128000
#define NP_   2048            // node dim padded to multiple of 32 (WMMA K)
#define TASK_ 64
#define ZD_   128             // TASK_ + CH_

typedef unsigned short u16;
typedef __attribute__((ext_vector_type(16))) __bf16 v16bf;
typedef __attribute__((ext_vector_type(8)))  float  v8f;

union FragU { v16bf v; uint4 q[2]; u16 h[16]; };

__device__ __forceinline__ u16 f2bf(float f) {
  union { float f; uint32_t u; } c; c.f = f;
  uint32_t u = c.u;
  return (u16)((u + 0x7FFFu + ((u >> 16) & 1u)) >> 16);  // RNE
}

// A-matrix 16x32 bf16 fragment (lane holds row m = lane&15):
// halves[0..7] = K(8*hi + 0..7), halves[8..15] = K(16 + 8*hi + 0..7)
__device__ __forceinline__ v16bf load_fragA(const u16* rowk, int hi) {
  FragU f;
  f.q[0] = *(const uint4*)(rowk + 8 * hi);
  f.q[1] = *(const uint4*)(rowk + 16 + 8 * hi);
  return f.v;
}
// B-matrix 32x16 bf16 fragment (lane holds column n = lane&15):
// halves[0..7] = K(8*hi + 0..7), halves[8..15] = K(16 + 8*hi + 0..7), memory layout [n][k] contiguous in k
__device__ __forceinline__ v16bf load_fragB(const u16* rowk, int hi) {
  FragU f;
  f.q[0] = *(const uint4*)(rowk + 8 * hi);
  f.q[1] = *(const uint4*)(rowk + 16 + 8 * hi);
  return f.v;
}
__device__ __forceinline__ v8f wmma_bf16(v16bf a, v16bf b, v8f c) {
  return __builtin_amdgcn_wmma_f32_16x16x32_bf16(false, a, false, b, (short)0, c,
                                                 false, false);
}

// ---------------- weight prep: f32 -> bf16 operand layouts ----------------
__global__ void k_prep(const float* filt_w, const float* gate_w, const float* skip_w,
                       const float* bn_g, u16* wf, u16* wg, u16* sw, float* bnsc) {
  for (int idx = threadIdx.x; idx < 3 * 64 * 128; idx += blockDim.x) {
    int i = idx / (64 * 128);
    int r = idx % (64 * 128);
    int co = r / 128, k = r % 128;
    int tap = k >> 6, ci = k & 63;
    size_t src = (((size_t)i * 64 + co) * 64 + ci) * 2 + tap;  // (3,CH,CH,2)
    wf[idx] = f2bf(filt_w[src]);
    wg[idx] = f2bf(gate_w[src]);
  }
  for (int idx = threadIdx.x; idx < 3 * 64 * 64; idx += blockDim.x)
    sw[idx] = f2bf(skip_w[idx]);                               // (3,CH,CH) = [i][co][ci]
  for (int idx = threadIdx.x; idx < 3 * 64; idx += blockDim.x)
    bnsc[idx] = bn_g[idx] * rsqrtf(1.0f + 1e-5f);
}

// ---------------- adp = softmax(relu(nv1@nv2), axis=1), stored transposed bf16 ----
__global__ void k_adp(const float* nv1, const float* nv2, u16* adpT) {
  __shared__ float rowbuf[N_];
  __shared__ float red[256];
  int v = blockIdx.x;
  float a[10];
#pragma unroll
  for (int k = 0; k < 10; ++k) a[k] = nv1[v * 10 + k];
  float lmax = -1e30f;
  for (int w = threadIdx.x; w < N_; w += blockDim.x) {
    float s = 0.f;
#pragma unroll
    for (int k = 0; k < 10; ++k) s += a[k] * nv2[k * N_ + w];
    s = fmaxf(s, 0.f);
    rowbuf[w] = s;
    lmax = fmaxf(lmax, s);
  }
  red[threadIdx.x] = lmax;
  __syncthreads();
  for (int off = 128; off > 0; off >>= 1) {
    if ((int)threadIdx.x < off) red[threadIdx.x] = fmaxf(red[threadIdx.x], red[threadIdx.x + off]);
    __syncthreads();
  }
  float m = red[0];
  __syncthreads();
  float lsum = 0.f;
  for (int w = threadIdx.x; w < N_; w += blockDim.x) {
    float e = __expf(rowbuf[w] - m);
    rowbuf[w] = e;
    lsum += e;
  }
  red[threadIdx.x] = lsum;
  __syncthreads();
  for (int off = 128; off > 0; off >>= 1) {
    if ((int)threadIdx.x < off) red[threadIdx.x] += red[threadIdx.x + off];
    __syncthreads();
  }
  float inv = 1.f / red[0];
  for (int w = threadIdx.x; w < N_; w += blockDim.x)
    adpT[(size_t)w * NP_ + v] = f2bf(rowbuf[w] * inv);  // adpT[w][v] = adp[v][w]
}

// ---------------- start 1x1 conv: X(B,T,N,F) -> x[(bn*T+t)*64+c] f32+bf16 --------
__global__ void k_start(const float* X, const float* sw, const float* sb,
                        float* xf, u16* xh) {
  size_t total = (size_t)BN_ * T_ * CH_;
  for (size_t idx = (size_t)blockIdx.x * blockDim.x + threadIdx.x; idx < total;
       idx += (size_t)gridDim.x * blockDim.x) {
    int c = (int)(idx & 63);
    size_t pos = idx >> 6;
    int t = (int)(pos % T_);
    size_t bn = pos / T_;
    int b = (int)(bn / N_), n = (int)(bn % N_);
    const float* xp = X + (((size_t)b * T_ + t) * N_ + n) * F_;
    const float* wp = sw + c * F_;
    float acc = sb[c];
#pragma unroll
    for (int k = 0; k < F_; ++k) acc += wp[k] * xp[k];
    xf[idx] = acc;
    xh[idx] = f2bf(acc);
  }
}

// -------- dilated gated conv layer as implicit WMMA GEMM + fused epilogue --------
// out[pos,co] over (bn,t), K = 2 taps x 64 ci. 1 block = 4 waves = 4 co-tiles.
__global__ void k_conv(const u16* xin_h, const float* xin_f,
                       float* xout_f, u16* xout_h, u16* fg_last,
                       const u16* wf, const u16* wg,
                       const float* fb, const float* gb,
                       const float* bnsc, const float* bnb,
                       int L_in, int L_out, int d, int M) {
  int lane = threadIdx.x & 31;
  int ntile = threadIdx.x >> 5;
  int mtile = blockIdx.x;
  int hi = lane >> 4, lo = lane & 15;
  int row = mtile * 16 + lo;
  int rpos = (row < M) ? row : 0;
  int bn = rpos / L_out, t = rpos - bn * L_out;
  const u16* arow0 = xin_h + ((size_t)bn * L_in + t) * 64;      // tap0
  const u16* arow1 = arow0 + (size_t)d * 64;                    // tap1 (t+d)
  const u16* browf = wf + (size_t)(ntile * 16 + lo) * 128;
  const u16* browg = wg + (size_t)(ntile * 16 + lo) * 128;
  v8f accf = {}; v8f accg = {};
#pragma unroll
  for (int kc = 0; kc < 4; ++kc) {
    const u16* ar = (kc < 2 ? arow0 : arow1) + (kc & 1) * 32;
    v16bf a   = load_fragA(ar, hi);
    v16bf bf_ = load_fragB(browf + kc * 32, hi);
    v16bf bg_ = load_fragB(browg + kc * 32, hi);
    accf = wmma_bf16(a, bf_, accf);
    accg = wmma_bf16(a, bg_, accg);
  }
  int co = ntile * 16 + lo;
  float fbv = fb[co], gbv = gb[co], sc = bnsc[co], bb = bnb[co];
#pragma unroll
  for (int r = 0; r < 8; ++r) {
    int m = mtile * 16 + r + 8 * hi;   // D element (M=r+8*hi, N=lo)
    if (m < M) {
      int bn2 = m / L_out, t2 = m - bn2 * L_out;
      float f = tanhf(accf[r] + fbv);
      float g = 1.f / (1.f + __expf(-(accg[r] + gbv)));
      float fg = f * g;                               // x = tanh(filt)*sigmoid(gate)
      if (t2 == L_out - 1) fg_last[(size_t)bn2 * 64 + co] = f2bf(fg);
      float res = xin_f[((size_t)bn2 * L_in + t2 + d) * 64 + co];
      float xo = (fg + res) * sc + bb;                // residual + eval BN
      size_t o = (size_t)m * 64 + co;
      xout_f[o] = xo;
      xout_h[o] = f2bf(xo);
    }
  }
}

// -------- skip 1x1 conv on last time position (WMMA, M=BN, N=64, K=64) ----------
__global__ void k_skip(const u16* fg_last, const u16* sw, const float* sb,
                       float* skip_acc, int init) {
  int lane = threadIdx.x & 31;
  int ntile = threadIdx.x >> 5;
  int mtile = blockIdx.x;
  int hi = lane >> 4, lo = lane & 15;
  const u16* arow = fg_last + (size_t)(mtile * 16 + lo) * 64;
  const u16* brow = sw + (size_t)(ntile * 16 + lo) * 64;
  v8f acc = {};
#pragma unroll
  for (int kc = 0; kc < 2; ++kc) {
    v16bf a = load_fragA(arow + kc * 32, hi);
    v16bf b = load_fragB(brow + kc * 32, hi);
    acc = wmma_bf16(a, b, acc);
  }
  int co = ntile * 16 + lo;
  float bias = sb[co];
#pragma unroll
  for (int r = 0; r < 8; ++r) {
    int m = mtile * 16 + r + 8 * hi;
    size_t o = (size_t)m * 64 + co;
    float v = acc[r] + bias;
    skip_acc[o] = init ? v : (skip_acc[o] + v);
  }
}

// -------- transpose skip (bn,c) -> skipT[b][c][v] bf16, zero-padded K -----------
__global__ void k_transpose(const float* skip_acc, u16* skipT) {
  size_t total = (size_t)B_ * 64 * NP_;
  for (size_t idx = (size_t)blockIdx.x * blockDim.x + threadIdx.x; idx < total;
       idx += (size_t)gridDim.x * blockDim.x) {
    int v = (int)(idx % NP_);
    size_t r = idx / NP_;
    int c = (int)(r % 64);
    int b = (int)(r / 64);
    float val = (v < N_) ? skip_acc[((size_t)b * N_ + v) * 64 + c] : 0.f;
    skipT[idx] = f2bf(val);
  }
}

// -------- batched WMMA GEMM: C[b] = A[b](64 x Kpad) * adpT^T -> (64 x 2000) -----
// Each wave owns a 16(M) x 64(N) tile: one A fragment feeds 4 independent WMMAs
// per K-step (4x A reuse, XDL latency hidden by independent accumulators).
__global__ void k_gemm(const u16* A, size_t a_bs, int lda,
                       const u16* Bm, int ldb,
                       float* Cf, size_t cf_bs, int ldc,
                       u16* Ch, size_t ch_bs,
                       int Mtiles, int Ngrp, int Ntiles, int Kch, int total) {
  int lane = threadIdx.x & 31;
  int wv = threadIdx.x >> 5;
  int task = blockIdx.x * 4 + wv;
  if (task >= total) return;                 // wave-uniform (EXEC stays full)
  int per_b = Mtiles * Ngrp;
  int b = task / per_b;
  int r = task - b * per_b;
  int mt = r / Ngrp, ng = r - mt * Ngrp;
  int hi = lane >> 4, lo = lane & 15;
  const u16* arow = A + (size_t)b * a_bs + (size_t)(mt * 16 + lo) * lda;
  const u16* brow[4];
  bool nvld[4];
#pragma unroll
  for (int j = 0; j < 4; ++j) {
    int nt = ng * 4 + j;
    nvld[j] = nt < Ntiles;                   // wave-uniform validity mask
    brow[j] = Bm + (size_t)((nvld[j] ? nt : 0) * 16 + lo) * ldb;
  }
  v8f acc[4] = {};
  for (int k = 0; k < Kch; ++k) {
    __builtin_prefetch(arow + (size_t)(k + 8) * 32, 0, 1);
    v16bf a = load_fragA(arow + (size_t)k * 32, hi);
#pragma unroll
    for (int j = 0; j < 4; ++j) {
      v16bf bb = load_fragB(brow[j] + (size_t)k * 32, hi);
      acc[j] = wmma_bf16(a, bb, acc[j]);
    }
  }
#pragma unroll
  for (int j = 0; j < 4; ++j) {
    if (!nvld[j]) continue;                  // wave-uniform
    int col = (ng * 4 + j) * 16 + lo;
#pragma unroll
    for (int rr = 0; rr < 8; ++rr) {
      int mm = mt * 16 + rr + 8 * hi;
      size_t o = (size_t)mm * ldc + col;
      Cf[(size_t)b * cf_bs + o] = acc[j][rr];
      if (Ch) Ch[(size_t)b * ch_bs + o] = f2bf(acc[j][rr]);
    }
  }
}

// -------- fused head: GCN 1x1 + task concat + 2 hypernets + BCE/MMD accum -------
__global__ void k_head(const float* skip_acc, const float* x1f, const float* x2f,
                       const float* gcn_w, const float* gcn_b, const float* task_emb,
                       const float* h0_w1, const float* h0_b1, const float* h0_w2,
                       const float* h0_b2, const float* h1_w1, const float* h1_b1,
                       const float* h1_w2, const float* h1_b2,
                       const int* C, const float* Y,
                       float* out, float* bce_acc, float* cnt, float* sums) {
  __shared__ float hb[4][192];
  __shared__ float zb[4][ZD_];
  __shared__ float hh[4][64];
  int lane = threadIdx.x & 31;
  int wv = threadIdx.x >> 5;
  int idx = blockIdx.x * 4 + wv;                 // (b,n) flat
  int b = idx / N_, n = idx % N_;
  for (int j = lane; j < 64; j += 32) {
    hb[wv][j]       = skip_acc[(size_t)idx * 64 + j];
    hb[wv][64 + j]  = x1f[((size_t)b * 64 + j) * NP_ + n];
    hb[wv][128 + j] = x2f[((size_t)b * 64 + j) * NP_ + n];
    zb[wv][j] = task_emb[j];
  }
  __syncthreads();
  for (int o = lane; o < 64; o += 32) {
    float acc = gcn_b[o];
    const float* wp = gcn_w + o * 192;
    for (int k = 0; k < 192; ++k) acc += wp[k] * hb[wv][k];
    zb[wv][64 + o] = fmaxf(acc, 0.f);            // gx (relu)
  }
  __syncthreads();
  float ys[2];
  for (int head = 0; head < 2; ++head) {
    const float* w1 = head ? h1_w1 : h0_w1;
    const float* b1 = head ? h1_b1 : h0_b1;
    const float* w2 = head ? h1_w2 : h0_w2;
    const float* b2 = head ? h1_b2 : h0_b2;
    for (int o = lane; o < 64; o += 32) {
      float acc = b1[o];
      const float* wp = w1 + o * ZD_;
      for (int k = 0; k < ZD_; ++k) acc += wp[k] * zb[wv][k];
      hh[wv][o] = fmaxf(acc, 0.f);
    }
    float part = hh[wv][lane] * w2[lane] + hh[wv][lane + 32] * w2[lane + 32];
    for (int off = 16; off > 0; off >>= 1) part += __shfl_xor(part, off, 32);
    ys[head] = 1.f / (1.f + __expf(-(part + b2[0])));
    __syncthreads();
  }
  int t = C[idx] > 0 ? 1 : 0;
  float y = t ? ys[1] : ys[0];
  if (lane == 0) {
    out[1 + idx] = y;
    out[1 + BN_ + idx] = ys[0];
    out[1 + 2 * BN_ + idx] = ys[1];
    float yc = fminf(fmaxf(y, 1e-7f), 1.f - 1e-7f);
    float yy = Y[idx];
    float bce = -(yy * __logf(yc) + (1.f - yy) * __logf(1.f - yc));
    atomicAdd(bce_acc, bce);
    atomicAdd(&cnt[t], 1.0f);
  }
  float* s = sums + (size_t)t * ZD_;             // [0]=control, [1]=treated
  for (int j = lane; j < ZD_; j += 32) atomicAdd(&s[j], zb[wv][j]);
}

// -------- final scalar loss: BCE mean + P1 * linear MMD ------------------------
__global__ void k_loss(const float* bce_acc, const float* cnt, const float* sums,
                       float* out) {
  __shared__ float red[ZD_];
  int j = threadIdx.x;
  float ct = fmaxf(cnt[1], 1.0f);
  float cc = fmaxf(cnt[0], 1.0f);
  float mt = sums[ZD_ + j] / ct;
  float mc = sums[j] / cc;
  float d = 2.0f * 0.5f * mt - 2.0f * 0.5f * mc;   // P_TREAT = 0.5
  red[j] = d * d;
  __syncthreads();
  for (int off = 64; off > 0; off >>= 1) {
    if (j < off) red[j] += red[j + off];
    __syncthreads();
  }
  if (j == 0) out[0] = bce_acc[0] / (float)BN_ + 1.0f * red[0];  // P1 = 1.0
}

// ==================================================================================
extern "C" void kernel_launch(void* const* d_in, const int* in_sizes, int n_in,
                              void* d_out, int out_size, void* d_ws, size_t ws_size,
                              hipStream_t stream) {
  (void)in_sizes; (void)n_in; (void)out_size; (void)ws_size;
  const float* X        = (const float*)d_in[0];
  const int*   C        = (const int*)  d_in[2];
  const float* Y        = (const float*)d_in[3];
  const float* start_w  = (const float*)d_in[5];
  const float* start_b  = (const float*)d_in[6];
  const float* filt_w   = (const float*)d_in[7];
  const float* filt_b   = (const float*)d_in[8];
  const float* gate_w   = (const float*)d_in[9];
  const float* gate_b   = (const float*)d_in[10];
  const float* skip_w   = (const float*)d_in[11];
  const float* skip_b   = (const float*)d_in[12];
  const float* bn_g     = (const float*)d_in[13];
  const float* bn_b     = (const float*)d_in[14];
  const float* nv1      = (const float*)d_in[15];
  const float* nv2      = (const float*)d_in[16];
  const float* task_emb = (const float*)d_in[17];
  const float* gcn_w    = (const float*)d_in[18];
  const float* gcn_b    = (const float*)d_in[19];
  const float* h0_w1    = (const float*)d_in[20];
  const float* h0_b1    = (const float*)d_in[21];
  const float* h0_w2    = (const float*)d_in[22];
  const float* h0_b2    = (const float*)d_in[23];
  const float* h1_w1    = (const float*)d_in[24];
  const float* h1_b1    = (const float*)d_in[25];
  const float* h1_w2    = (const float*)d_in[26];
  const float* h1_b2    = (const float*)d_in[27];
  float* out = (float*)d_out;

  char* cur = (char*)d_ws;
  auto alloc = [&](size_t bytes) -> char* {
    char* p = cur;
    cur += (bytes + 255) & ~(size_t)255;
    return p;
  };
  // ping-pong activation buffers (buf0 sized L=8, buf1 sized L=7)
  float* b0f = (float*)alloc((size_t)BN_ * 8 * 64 * 4);
  u16*   b0h = (u16*)  alloc((size_t)BN_ * 8 * 64 * 2);
  float* b1f = (float*)alloc((size_t)BN_ * 7 * 64 * 4);
  u16*   b1h = (u16*)  alloc((size_t)BN_ * 7 * 64 * 2);
  u16* adpT = (u16*)alloc((size_t)N_ * NP_ * 2);
  u16* wfb  = (u16*)alloc((size_t)3 * 64 * 128 * 2);
  u16* wgb  = (u16*)alloc((size_t)3 * 64 * 128 * 2);
  u16* swb  = (u16*)alloc((size_t)3 * 64 * 64 * 2);
  float* bnsc = (float*)alloc((size_t)3 * 64 * 4);
  u16* fg_last   = (u16*)  alloc((size_t)BN_ * 64 * 2);
  float* skip_acc= (float*)alloc((size_t)BN_ * 64 * 4);
  u16* skipT     = (u16*)  alloc((size_t)B_ * 64 * NP_ * 2);
  float* g1f     = (float*)alloc((size_t)B_ * 64 * NP_ * 4);
  u16*   g1h     = (u16*)  alloc((size_t)B_ * 64 * NP_ * 2);
  float* g2f     = (float*)alloc((size_t)B_ * 64 * NP_ * 4);
  float* bce_acc = (float*)alloc(4);
  float* cnt     = (float*)alloc(2 * 4);
  float* sums    = (float*)alloc((size_t)2 * ZD_ * 4);

  // zero-init pads & accumulators (graph-capturable stream ops)
  hipMemsetAsync(adpT, 0, (size_t)N_ * NP_ * 2, stream);
  hipMemsetAsync(g1h, 0, (size_t)B_ * 64 * NP_ * 2, stream);
  hipMemsetAsync(bce_acc, 0, 4, stream);
  hipMemsetAsync(cnt, 0, 8, stream);
  hipMemsetAsync(sums, 0, (size_t)2 * ZD_ * 4, stream);

  k_prep<<<1, 256, 0, stream>>>(filt_w, gate_w, skip_w, bn_g, wfb, wgb, swb, bnsc);
  k_adp<<<N_, 256, 0, stream>>>(nv1, nv2, adpT);
  {
    size_t tot = (size_t)BN_ * T_ * CH_;
    k_start<<<(int)((tot + 255) / 256), 256, 0, stream>>>(X, start_w, start_b, b0f, b0h);
  }
  // layer 0: L 8 -> 7, d=1
  k_conv<<<(BN_ * 7) / 16, 128, 0, stream>>>(b0h, b0f, b1f, b1h, fg_last,
      wfb, wgb, filt_b, gate_b, bnsc, bn_b, 8, 7, 1, BN_ * 7);
  k_skip<<<BN_ / 16, 128, 0, stream>>>(fg_last, swb, skip_b, skip_acc, 1);
  // layer 1: L 7 -> 5, d=2
  k_conv<<<(BN_ * 5) / 16, 128, 0, stream>>>(b1h, b1f, b0f, b0h, fg_last,
      wfb + 64 * 128, wgb + 64 * 128, filt_b + 64, gate_b + 64, bnsc + 64, bn_b + 64,
      7, 5, 2, BN_ * 5);
  k_skip<<<BN_ / 16, 128, 0, stream>>>(fg_last, swb + 64 * 64, skip_b + 64, skip_acc, 0);
  // layer 2: L 5 -> 1, d=4
  k_conv<<<BN_ / 16, 128, 0, stream>>>(b0h, b0f, b1f, b1h, fg_last,
      wfb + 2 * 64 * 128, wgb + 2 * 64 * 128, filt_b + 128, gate_b + 128, bnsc + 128,
      bn_b + 128, 5, 1, 4, BN_);
  k_skip<<<BN_ / 16, 128, 0, stream>>>(fg_last, swb + 2 * 64 * 64, skip_b + 128, skip_acc, 0);

  {
    size_t tot = (size_t)B_ * 64 * NP_;
    k_transpose<<<(int)((tot + 255) / 256), 256, 0, stream>>>(skip_acc, skipT);
  }
  // GCN GEMM1: x1 = skip @ adp ; GEMM2: x2 = x1 @ adp
  // Mtiles=4, Ngrp=32 (4 N-tiles per wave, 125 valid), Kch=64
  int total_t = B_ * 4 * 32;                      // 8192 wave tasks
  k_gemm<<<total_t / 4, 128, 0, stream>>>(skipT, (size_t)64 * NP_, NP_, adpT, NP_,
      g1f, (size_t)64 * NP_, NP_, g1h, (size_t)64 * NP_, 4, 32, 125, 64, total_t);
  k_gemm<<<total_t / 4, 128, 0, stream>>>(g1h, (size_t)64 * NP_, NP_, adpT, NP_,
      g2f, (size_t)64 * NP_, NP_, (u16*)nullptr, 0, 4, 32, 125, 64, total_t);

  k_head<<<BN_ / 4, 128, 0, stream>>>(skip_acc, g1f, g2f, gcn_w, gcn_b, task_emb,
      h0_w1, h0_b1, h0_w2, h0_b2, h1_w1, h1_b1, h1_w2, h1_b2, C, Y,
      out, bce_acc, cnt, sums);
  k_loss<<<1, ZD_, 0, stream>>>(bce_acc, cnt, sums, out);
}
